// TreeLayer_30777735643844
// MI455X (gfx1250) — matrix operand order repaired and checked by reading, not compiled, not run on hardware
//
#include <hip/hip_runtime.h>
#include <hip/hip_bf16.h>

typedef __bf16 bf16_t;
typedef __attribute__((ext_vector_type(16))) __bf16 v16bf;
typedef __attribute__((ext_vector_type(8)))  __bf16 v8bf;
typedef __attribute__((ext_vector_type(8)))  float  v8f;

#define BATCH 16
#define SEQ   2048
#define HD    512
#define ROWS  (BATCH * SEQ)      // 32768 flattened rows for the shared-weight GEMMs

static __device__ inline v8f zero8() {
  v8f z;
#pragma unroll
  for (int i = 0; i < 8; ++i) z[i] = 0.0f;
  return z;
}

// A-matrix fragment (16x32 bf16): lane l holds row (l&15); K = 2*kb..2*kb+15
// contiguous where kb = (lane<16 ? 0 : 8). Two contiguous 16B LDS loads.
static __device__ inline v16bf load_a_frag(const bf16_t* row, int kb) {
  v8bf lo = *(const v8bf*)(row + 2 * kb);
  v8bf hi = *(const v8bf*)(row + 2 * kb + 8);
  return __builtin_shufflevector(lo, hi, 0, 1, 2, 3, 4, 5, 6, 7,
                                 8, 9, 10, 11, 12, 13, 14, 15);
}
// B-matrix fragment (32x16 bf16): lane l holds column (l&15), K = kbB..kbB+15
// contiguous (kbB = lane<16 ? 0 : 16). Single contiguous 32B LDS load.
static __device__ inline v16bf load_b_frag(const bf16_t* p) {
  return *(const v16bf*)p;
}
static __device__ inline v8f wmma_bf16(v16bf a, v16bf b, v8f c) {
  return __builtin_amdgcn_wmma_f32_16x16x32_bf16(false, a, false, b,
                                                 (short)0, c, false, false);
}

// ---------------------------------------------------------------------------
// f32 -> bf16 conversion (grid-stride)
// ---------------------------------------------------------------------------
__global__ void f32_to_bf16_kernel(const float* __restrict__ s,
                                   bf16_t* __restrict__ d, int n) {
  int i = blockIdx.x * blockDim.x + threadIdx.x;
  int stride = gridDim.x * blockDim.x;
  for (; i < n; i += stride) d[i] = (bf16_t)s[i];
}

// ---------------------------------------------------------------------------
// C = relu(X @ W + bias) ; X:[32768,512] bf16, W:[512,512] bf16 (row-major,
// [in,out]).  Block = 128x128 tile, 256 threads = 8 waves, wave tile = 32x64.
// K staged 64-deep per barrier pair -> 16 WMMAs per stage.
// F32OUT selects branch-free f32 vs bf16 epilogue at compile time.
// ---------------------------------------------------------------------------
template <bool F32OUT>
__global__ __launch_bounds__(256) void gemm_bias_relu_kernel(
    const bf16_t* __restrict__ X, const bf16_t* __restrict__ W,
    const float* __restrict__ bias, bf16_t* __restrict__ outb,
    float* __restrict__ outf) {
  __shared__ __align__(16) bf16_t As[128 * 64];   // [row][k]     16 KB
  __shared__ __align__(16) bf16_t Bt[128 * 64];   // [col][k]     16 KB

  const int tid  = threadIdx.x;
  const int lane = tid & 31;
  const int wave = tid >> 5;
  const int r0 = blockIdx.x * 128;
  const int c0 = blockIdx.y * 128;
  const int wr = wave & 3;    // 4 waves down rows
  const int wc = wave >> 2;   // 2 waves across cols
  const int cl  = lane & 15;
  const int kb  = (lane < 16) ? 0 : 8;
  const int kbB = (lane < 16) ? 0 : 16;
  const int hi8 = (lane < 16) ? 0 : 8;

  v8f acc[2][4];
#pragma unroll
  for (int m = 0; m < 2; ++m)
#pragma unroll
    for (int n = 0; n < 4; ++n) acc[m][n] = zero8();

  for (int k0 = 0; k0 < HD; k0 += 64) {
    __syncthreads();
    // Stage A tile: 128x64, contiguous 16B chunks
#pragma unroll
    for (int it = 0; it < 4; ++it) {
      int idx = tid + it * 256;            // 1024 chunks of 8 elems
      int r = idx >> 3;
      int kk = (idx & 7) * 8;
      *(v8bf*)(As + r * 64 + kk) =
          *(const v8bf*)(X + (size_t)(r0 + r) * HD + k0 + kk);
    }
    // Stage B tile transposed to [col][k] so B fragments are contiguous
#pragma unroll
    for (int it = 0; it < 4; ++it) {
      int idx = tid + it * 256;
      int kk = idx >> 4;                   // 0..63
      int cb = (idx & 15) * 8;
      v8bf d = *(const v8bf*)(W + (size_t)(k0 + kk) * HD + c0 + cb);
#pragma unroll
      for (int j = 0; j < 8; ++j) Bt[(cb + j) * 64 + kk] = d[j];
    }
    // Prefetch next K stage into cache while we do WMMAs on this one
    if (k0 + 64 < HD) {
      __builtin_prefetch(X + (size_t)(r0 + (tid >> 1)) * HD + k0 + 64 +
                             (tid & 1) * 32, 0, 0);
      __builtin_prefetch(W + (size_t)(k0 + 64 + (tid >> 2)) * HD + c0 +
                             (tid & 3) * 32, 0, 0);
    }
    __syncthreads();

#pragma unroll
    for (int kk = 0; kk < 64; kk += 32) {
      v16bf a[2], b[4];
#pragma unroll
      for (int m = 0; m < 2; ++m)
        a[m] = load_a_frag(As + (wr * 32 + m * 16 + cl) * 64 + kk, kb);
#pragma unroll
      for (int n = 0; n < 4; ++n)
        b[n] = load_b_frag(Bt + (wc * 64 + n * 16 + cl) * 64 + kk + kbB);
#pragma unroll
      for (int m = 0; m < 2; ++m)
#pragma unroll
        for (int n = 0; n < 4; ++n)
          acc[m][n] = wmma_bf16(a[m], b[n], acc[m][n]);
    }
  }

  // Epilogue: bias + ReLU, branch-free store (type chosen at compile time)
#pragma unroll
  for (int n = 0; n < 4; ++n) {
    int cg = c0 + wc * 64 + n * 16 + cl;
    float bs = bias[cg];
#pragma unroll
    for (int m = 0; m < 2; ++m) {
#pragma unroll
      for (int vr = 0; vr < 8; ++vr) {
        int rg = r0 + wr * 32 + m * 16 + vr + hi8;
        float v = fmaxf(acc[m][n][vr] + bs, 0.0f);
        size_t gi = (size_t)rg * HD + cg;
        if constexpr (F32OUT) outf[gi] = v;
        else                  outb[gi] = (bf16_t)v;
      }
    }
  }
}

// ---------------------------------------------------------------------------
// Flash attention with elementwise bias:  O = softmax((QK^T) * -A) @ V + V
// Block = 16 query rows of one batch, 128 threads (4 waves).
// Wave0 computes 16x32 score tiles + online softmax; every wave accumulates a
// 128-wide H slice of P@V.  K chunk and (transposed) V chunk share one buffer.
// ---------------------------------------------------------------------------
__global__ __launch_bounds__(128) void attention_kernel(
    const bf16_t* __restrict__ Q, const bf16_t* __restrict__ K,
    const bf16_t* __restrict__ V, const float* __restrict__ A,
    bf16_t* __restrict__ O) {
  __shared__ __align__(16) bf16_t Qs[16 * HD];    // 16 KB, resident
  __shared__ __align__(16) bf16_t KV[32 * HD];    // 32 KB: K [key][h], then V^T [h][key]
  __shared__ __align__(16) bf16_t Pb[16 * 32];    // probabilities, bf16
  __shared__ float Mrow[16], Srow[16], Scl[16], Inv[16], Tmx[16], Tsm[16];

  const int tid  = threadIdx.x;
  const int lane = tid & 31;
  const int wave = tid >> 5;
  const int b     = blockIdx.x >> 7;           // SEQ/16 == 128 blocks per batch
  const int qbase = (blockIdx.x & 127) * 16;
  const int cl  = lane & 15;
  const int kb  = (lane < 16) ? 0 : 8;
  const int kbB = (lane < 16) ? 0 : 16;
  const int hi8 = (lane < 16) ? 0 : 8;

  const size_t rowbase = (size_t)b * SEQ;
  const float* Abase = A + (size_t)b * SEQ * SEQ;

  // Load the 16 query rows once
#pragma unroll
  for (int it = 0; it < 8; ++it) {
    int idx = tid + it * 128;                  // 1024 chunks of 8 elems
    int r = idx >> 6;
    int hb = (idx & 63) * 8;
    *(v8bf*)(Qs + r * HD + hb) =
        *(const v8bf*)(Q + (rowbase + qbase + r) * HD + hb);
  }
  if (tid < 16) { Mrow[tid] = -__builtin_inff(); Srow[tid] = 0.0f; }

  v8f acc[8];
#pragma unroll
  for (int t = 0; t < 8; ++t) acc[t] = zero8();

  for (int kc = 0; kc < SEQ; kc += 32) {
    __syncthreads();
    // Prefetch next chunk's K/V rows and this chunk's A bias rows
    if (kc + 32 < SEQ) {
      int pr = tid >> 2;                       // 32 key rows
      int ph = (tid & 3) * 128;
      __builtin_prefetch(K + (rowbase + kc + 32 + pr) * HD + ph, 0, 0);
      __builtin_prefetch(V + (rowbase + kc + 32 + pr) * HD + ph, 0, 0);
    }
    if (tid < 16)
      __builtin_prefetch(Abase + (size_t)(qbase + tid) * SEQ + kc, 0, 0);

    // --- Stage K chunk: [key][h] ---
#pragma unroll
    for (int it = 0; it < 16; ++it) {
      int idx = tid + it * 128;                // 2048 chunks of 8
      int key = idx >> 6;
      int hb = (idx & 63) * 8;
      *(v8bf*)(KV + key * HD + hb) =
          *(const v8bf*)(K + (rowbase + kc + key) * HD + hb);
    }
    __syncthreads();

    if (wave == 0) {
      // S[16 x 32] = Q @ K^T over H=512 (16 k-steps, 2 key subtiles)
      v8f s0 = zero8(), s1 = zero8();
      const bf16_t* qrow = Qs + cl * HD;
      for (int hb = 0; hb < HD; hb += 32) {
        v16bf aq = load_a_frag(qrow + hb, kb);
        v16bf b0 = load_b_frag(KV + (size_t)cl * HD + hb + kbB);
        v16bf b1 = load_b_frag(KV + (size_t)(16 + cl) * HD + hb + kbB);
        s0 = wmma_bf16(aq, b0, s0);
        s1 = wmma_bf16(aq, b1, s1);
      }
      // scores *= -A  (elementwise bias from HBM)
#pragma unroll
      for (int vr = 0; vr < 8; ++vr) {
        const float* Ar = Abase + (size_t)(qbase + vr + hi8) * SEQ + kc;
        s0[vr] *= -Ar[cl];
        s1[vr] *= -Ar[16 + cl];
      }
      // chunk row-max via width-16 shuffles
#pragma unroll
      for (int vr = 0; vr < 8; ++vr) {
        float m = fmaxf(s0[vr], s1[vr]);
        for (int off = 8; off >= 1; off >>= 1)
          m = fmaxf(m, __shfl_xor(m, off, 16));
        if (cl == 0) Tmx[vr + hi8] = m;
      }
      __threadfence_block();
      if (lane < 16) {
        float mo = Mrow[lane];
        float mn = fmaxf(mo, Tmx[lane]);
        Scl[lane] = __expf(mo - mn);           // rescale for running state
        Mrow[lane] = mn;
      }
      __threadfence_block();
      // P = exp(s - max), row sums, bf16 P tile for the P@V WMMAs
#pragma unroll
      for (int vr = 0; vr < 8; ++vr) {
        float mn = Mrow[vr + hi8];
        float p0 = __expf(s0[vr] - mn);
        float p1 = __expf(s1[vr] - mn);
        Pb[(vr + hi8) * 32 + cl]      = (bf16_t)p0;
        Pb[(vr + hi8) * 32 + 16 + cl] = (bf16_t)p1;
        float rs = p0 + p1;
        for (int off = 8; off >= 1; off >>= 1) rs += __shfl_xor(rs, off, 16);
        if (cl == 0) Tsm[vr + hi8] = rs;
      }
      __threadfence_block();
      if (lane < 16) Srow[lane] = Srow[lane] * Scl[lane] + Tsm[lane];
    }
    __syncthreads();   // K dead; P/Scl ready

    // --- Stage V chunk transposed: [h][key] (reuses the K buffer) ---
#pragma unroll
    for (int it = 0; it < 16; ++it) {
      int idx = tid + it * 128;
      int key = idx >> 6;
      int hb = (idx & 63) * 8;
      v8bf d = *(const v8bf*)(V + (rowbase + kc + key) * HD + hb);
#pragma unroll
      for (int j = 0; j < 8; ++j) KV[(hb + j) * 32 + key] = d[j];
    }
    __syncthreads();

    // --- Every wave: rescale accumulators, acc += P @ V_slice ---
    {
      v16bf pa = load_a_frag(Pb + cl * 32, kb);
      float scl[8];
#pragma unroll
      for (int vr = 0; vr < 8; ++vr) scl[vr] = Scl[vr + hi8];
#pragma unroll
      for (int t = 0; t < 8; ++t) {
        int hcol = wave * 128 + t * 16 + cl;
        v16bf bv = load_b_frag(KV + (size_t)hcol * 32 + kbB);
        v8f c = acc[t];
#pragma unroll
        for (int vr = 0; vr < 8; ++vr) c[vr] *= scl[vr];
        acc[t] = wmma_bf16(pa, bv, c);
      }
    }
  }

  if (tid < 16) Inv[tid] = 1.0f / Srow[tid];
  __syncthreads();

  // O = acc/rowsum + V (residual), store bf16 for the final GEMM
#pragma unroll
  for (int t = 0; t < 8; ++t) {
    int cg = wave * 128 + t * 16 + cl;
#pragma unroll
    for (int vr = 0; vr < 8; ++vr) {
      size_t gi = (rowbase + qbase + vr + hi8) * HD + cg;
      float o = acc[t][vr] * Inv[vr + hi8] + (float)V[gi];
      O[gi] = (bf16_t)o;
    }
  }
}

// ---------------------------------------------------------------------------
extern "C" void kernel_launch(void* const* d_in, const int* in_sizes, int n_in,
                              void* d_out, int out_size, void* d_ws,
                              size_t ws_size, hipStream_t stream) {
  const float* A    = (const float*)d_in[0];
  const float* h    = (const float*)d_in[1];
  const float* Wv1  = (const float*)d_in[2];
  const float* bv1  = (const float*)d_in[3];
  const float* Wv2  = (const float*)d_in[4];
  const float* bv2  = (const float*)d_in[5];
  const float* Wk   = (const float*)d_in[6];
  const float* bk   = (const float*)d_in[7];
  const float* Wq   = (const float*)d_in[8];
  const float* bq   = (const float*)d_in[9];
  const float* Wout = (const float*)d_in[10];
  const float* bout = (const float*)d_in[11];
  float* out = (float*)d_out;

  const size_t act = (size_t)ROWS * HD;         // 16.7M elems
  const size_t wsz = (size_t)HD * HD;           // 262144 elems
  char* w = (char*)d_ws;
  bf16_t* hb    = (bf16_t*)w; w += act * 2;
  bf16_t* qbuf  = (bf16_t*)w; w += act * 2;
  bf16_t* kbuf  = (bf16_t*)w; w += act * 2;
  bf16_t* vbuf  = (bf16_t*)w; w += act * 2;
  bf16_t* tbuf  = (bf16_t*)w; w += act * 2;     // v1, then reused for att out
  bf16_t* wv1b  = (bf16_t*)w; w += wsz * 2;
  bf16_t* wv2b  = (bf16_t*)w; w += wsz * 2;
  bf16_t* wkb   = (bf16_t*)w; w += wsz * 2;
  bf16_t* wqb   = (bf16_t*)w; w += wsz * 2;
  bf16_t* woutb = (bf16_t*)w; w += wsz * 2;

  f32_to_bf16_kernel<<<2048, 256, 0, stream>>>(h, hb, (int)act);
  f32_to_bf16_kernel<<<256, 256, 0, stream>>>(Wv1, wv1b, (int)wsz);
  f32_to_bf16_kernel<<<256, 256, 0, stream>>>(Wv2, wv2b, (int)wsz);
  f32_to_bf16_kernel<<<256, 256, 0, stream>>>(Wk, wkb, (int)wsz);
  f32_to_bf16_kernel<<<256, 256, 0, stream>>>(Wq, wqb, (int)wsz);
  f32_to_bf16_kernel<<<256, 256, 0, stream>>>(Wout, woutb, (int)wsz);

  dim3 gg(ROWS / 128, HD / 128);                 // 256 x 4 blocks
  // t1 = relu(h @ Wv1 + bv1)
  gemm_bias_relu_kernel<false><<<gg, 256, 0, stream>>>(hb, wv1b, bv1, tbuf,
                                                       nullptr);
  // v = relu(t1 @ Wv2 + bv2)
  gemm_bias_relu_kernel<false><<<gg, 256, 0, stream>>>(tbuf, wv2b, bv2, vbuf,
                                                       nullptr);
  // q = relu(h @ Wq + bq),  k = relu(h @ Wk + bk)
  gemm_bias_relu_kernel<false><<<gg, 256, 0, stream>>>(hb, wqb, bq, qbuf,
                                                       nullptr);
  gemm_bias_relu_kernel<false><<<gg, 256, 0, stream>>>(hb, wkb, bk, kbuf,
                                                       nullptr);
  // att_out = softmax(qk^T * -A) @ v + v   (tbuf reused as output)
  attention_kernel<<<BATCH * (SEQ / 16), 128, 0, stream>>>(qbuf, kbuf, vbuf, A,
                                                           tbuf);
  // out = relu(att_out @ Wout + bout)  -> f32
  gemm_bias_relu_kernel<true><<<gg, 256, 0, stream>>>(tbuf, woutb, bout,
                                                      nullptr, out);
}
